// TritonDynamicLinear_71760313581857
// MI455X (gfx1250) — matrix operand order repaired and checked by reading, not compiled, not run on hardware
//
#include <hip/hip_runtime.h>

typedef __attribute__((ext_vector_type(16))) __bf16 v16bf;
typedef __attribute__((ext_vector_type(8)))  float  v8f;

#define MDIM 4096
#define NDIM 4096
#define KDIM 4096
#define NH 128          // N / 32 block rows of weight
#define NW 128          // K / 32 block cols of weight

#define BM 256
#define BN 128
#define BK 32
#define LDT 40          // padded LDS row stride (bf16 elems): 32 + 8 pad, keeps 16B align

union FragU {
    uint4 q[2];
    v16bf v;
};

// ---------------------------------------------------------------------------
// Kernel 1: per-32x32-block sum of int mask, threshold with conv_bias -> int 0/1
// ---------------------------------------------------------------------------
__global__ __launch_bounds__(256) void block_mask_kernel(
    const int* __restrict__ mask, const float* __restrict__ conv_bias,
    int* __restrict__ bmask) {
    const int blk = blockIdx.x;          // 0 .. NH*NW-1
    const int bh = blk >> 7;             // n-block
    const int bw = blk & 127;            // k-block
    const int tid = threadIdx.x;
    const int row = tid >> 3;            // 0..31
    const int cg  = tid & 7;             // 0..7 -> 4 ints each

    const int* p = mask + (size_t)(bh * 32 + row) * KDIM + bw * 32 + cg * 4;
    int4 v = *(const int4*)p;
    int s = v.x + v.y + v.z + v.w;

    // wave32 reduction
    for (int o = 16; o > 0; o >>= 1) s += __shfl_down(s, o, 32);

    __shared__ int wsum[8];
    const int wave = tid >> 5, lane = tid & 31;
    if (lane == 0) wsum[wave] = s;
    __syncthreads();
    if (tid == 0) {
        int t = 0;
        #pragma unroll
        for (int i = 0; i < 8; ++i) t += wsum[i];
        bmask[blk] = ((float)t + conv_bias[0] > 0.0f) ? 1 : 0;
    }
}

// ---------------------------------------------------------------------------
// Kernel 2: fp32 -> bf16 (data)
// ---------------------------------------------------------------------------
__global__ __launch_bounds__(256) void cvt_data_kernel(
    const float* __restrict__ in, __bf16* __restrict__ out) {
    const size_t i = ((size_t)blockIdx.x * 256 + threadIdx.x) * 4;
    float4 f = *(const float4*)(in + i);
    union { __bf16 h[4]; uint2 u; } pk;
    pk.h[0] = (__bf16)f.x; pk.h[1] = (__bf16)f.y;
    pk.h[2] = (__bf16)f.z; pk.h[3] = (__bf16)f.w;
    *(uint2*)(out + i) = pk.u;
}

// ---------------------------------------------------------------------------
// Kernel 3: fp32 -> bf16 with 32x32 block mask (weight)
// ---------------------------------------------------------------------------
__global__ __launch_bounds__(256) void cvt_weight_kernel(
    const float* __restrict__ w, const int* __restrict__ bmask,
    __bf16* __restrict__ out) {
    const size_t i = ((size_t)blockIdx.x * 256 + threadIdx.x) * 4;
    const int n = (int)(i >> 12);        // row (N)
    const int k = (int)(i & 4095);       // col (K); 4 elems stay in one block
    const float m = bmask[(n >> 5) * NW + (k >> 5)] ? 1.0f : 0.0f;
    float4 f = *(const float4*)(w + i);
    union { __bf16 h[4]; uint2 u; } pk;
    pk.h[0] = (__bf16)(f.x * m); pk.h[1] = (__bf16)(f.y * m);
    pk.h[2] = (__bf16)(f.z * m); pk.h[3] = (__bf16)(f.w * m);
    *(uint2*)(out + i) = pk.u;
}

// ---------------------------------------------------------------------------
// Kernel 4: bf16 WMMA GEMM  out[M,N] = A[M,K] * W[N,K]^T + bias
// 256x128 tile / workgroup, 8 waves (4Mx2N), each wave 64x64 = 4x4 WMMA accs.
// Block-sparse: scalar-branch skip of inactive 32x32 weight blocks.
// ---------------------------------------------------------------------------
__global__ __launch_bounds__(256) void gemm_bf16_wmma_kernel(
    const __bf16* __restrict__ A, const __bf16* __restrict__ W,
    const float* __restrict__ bias, const int* __restrict__ bmask,
    float* __restrict__ out) {
    __shared__ __bf16 sA[2][BM * LDT];   // 2 * 256*40 * 2B = 40 KB
    __shared__ __bf16 sB[2][BN * LDT];   // 2 * 128*40 * 2B = 20 KB

    const int tid  = threadIdx.x;
    const int lane = tid & 31;
    const int wave = tid >> 5;
    const int m0 = blockIdx.y * BM;
    const int n0 = blockIdx.x * BN;
    const int wm = (wave >> 1) * 64;     // wave M offset: 0/64/128/192
    const int wn = (wave & 1) * 64;      // wave N offset: 0/64

    // staging map: thread -> (row 0..63, 16B column segment)
    const int srow = tid >> 2;           // 0..63
    const int scol = (tid & 3) * 8;      // 0,8,16,24 (bf16 elems)

    const __bf16* gA = A + (size_t)(m0 + srow) * KDIM + scol;
    const __bf16* gB = W + (size_t)(n0 + srow) * KDIM + scol;
    const size_t rq = (size_t)64 * KDIM; // 64-row stride in elements

    // block-mask rows for this wave's two N-blocks
    const int nb0 = (n0 + wn) >> 5;
    const int* bm0p = bmask + (size_t)nb0 * NW;
    const int* bm1p = bm0p + NW;

    v8f acc[4][4] = {};

    // fragment addressing (per ISA bf16 VGPR layouts)
    const int arow0 = (wm + (lane & 15)) * LDT;          // A: M = lane%16 (+mi*16)
    const int akb   = (lane < 16) ? 0 : 8;               // A: K 0..7 / 8..15 (+16 elems)
    const int brow0 = (wn + (lane & 15)) * LDT;          // B: N = lane%16 (+ni*16)
    const int bkb   = (lane < 16) ? 0 : 8;               // B: K 0..7 / 8..15 (+16 elems)

    // preload k-tile 0 into LDS buffer 0
    {
        #pragma unroll
        for (int i = 0; i < 4; ++i)
            *(uint4*)&sA[0][(srow + i * 64) * LDT + scol] =
                *(const uint4*)(gA + (size_t)i * rq);
        #pragma unroll
        for (int i = 0; i < 2; ++i)
            *(uint4*)&sB[0][(srow + i * 64) * LDT + scol] =
                *(const uint4*)(gB + (size_t)i * rq);
    }
    __syncthreads();

    int curM0 = bm0p[0];
    int curM1 = bm1p[0];

    const int nkt = KDIM / BK;           // 128
    int buf = 0;
    for (int kt = 0; kt < nkt; ++kt) {
        // prefetch next k-tile + next mask words into registers
        uint4 na[4], nbq[2];
        int nM0 = 0, nM1 = 0;
        const bool has_next = (kt + 1) < nkt;
        if (has_next) {
            const size_t go = (size_t)(kt + 1) * BK;
            #pragma unroll
            for (int i = 0; i < 4; ++i)
                na[i] = *(const uint4*)(gA + go + (size_t)i * rq);
            #pragma unroll
            for (int i = 0; i < 2; ++i)
                nbq[i] = *(const uint4*)(gB + go + (size_t)i * rq);
            nM0 = bm0p[kt + 1];
            nM1 = bm1p[kt + 1];
        }

        // scalarize wave-uniform block masks -> scalar branch, EXEC untouched
        const int s0 = __builtin_amdgcn_readfirstlane(curM0);
        const int s1 = __builtin_amdgcn_readfirstlane(curM1);

        if (s0 | s1) {
            FragU af[4];
            #pragma unroll
            for (int mi = 0; mi < 4; ++mi) {
                const int base = arow0 + mi * 16 * LDT + akb;
                af[mi].q[0] = *(const uint4*)&sA[buf][base];
                af[mi].q[1] = *(const uint4*)&sA[buf][base + 16];
            }
            if (s0) {
                FragU bf[2];
                #pragma unroll
                for (int ni = 0; ni < 2; ++ni) {
                    const int base = brow0 + ni * 16 * LDT + bkb;
                    bf[ni].q[0] = *(const uint4*)&sB[buf][base];
                    bf[ni].q[1] = *(const uint4*)&sB[buf][base + 16];
                }
                #pragma unroll
                for (int ni = 0; ni < 2; ++ni)
                    #pragma unroll
                    for (int mi = 0; mi < 4; ++mi)
                        acc[mi][ni] = __builtin_amdgcn_wmma_f32_16x16x32_bf16(
                            false, af[mi].v, false, bf[ni].v,
                            (short)0, acc[mi][ni], false, false);
            }
            if (s1) {
                FragU bf[2];
                #pragma unroll
                for (int i = 0; i < 2; ++i) {
                    const int base = brow0 + (2 + i) * 16 * LDT + bkb;
                    bf[i].q[0] = *(const uint4*)&sB[buf][base];
                    bf[i].q[1] = *(const uint4*)&sB[buf][base + 16];
                }
                #pragma unroll
                for (int i = 0; i < 2; ++i)
                    #pragma unroll
                    for (int mi = 0; mi < 4; ++mi)
                        acc[mi][2 + i] = __builtin_amdgcn_wmma_f32_16x16x32_bf16(
                            false, af[mi].v, false, bf[i].v,
                            (short)0, acc[mi][2 + i], false, false);
            }
        }

        if (has_next) {
            const int nb = buf ^ 1;      // write other buffer; no race with readers
            #pragma unroll
            for (int i = 0; i < 4; ++i)
                *(uint4*)&sA[nb][(srow + i * 64) * LDT + scol] = na[i];
            #pragma unroll
            for (int i = 0; i < 2; ++i)
                *(uint4*)&sB[nb][(srow + i * 64) * LDT + scol] = nbq[i];
            __syncthreads();
            buf = nb;
            curM0 = nM0;
            curM1 = nM1;
        }
    }

    // epilogue: C layout — VGPR r, lanes0-15: M=r, N=lane; lanes16-31: M=8+r
    #pragma unroll
    for (int mi = 0; mi < 4; ++mi) {
        const int mbase = m0 + wm + mi * 16 + ((lane >> 4) ? 8 : 0);
        #pragma unroll
        for (int ni = 0; ni < 4; ++ni) {
            const int n = n0 + wn + ni * 16 + (lane & 15);
            const float bv = bias[n];
            #pragma unroll
            for (int r = 0; r < 8; ++r)
                out[(size_t)(mbase + r) * NDIM + n] = acc[mi][ni][r] + bv;
        }
    }
}

// ---------------------------------------------------------------------------
extern "C" void kernel_launch(void* const* d_in, const int* in_sizes, int n_in,
                              void* d_out, int out_size, void* d_ws, size_t ws_size,
                              hipStream_t stream) {
    const float* data      = (const float*)d_in[0];
    const float* weight    = (const float*)d_in[1];
    const float* bias      = (const float*)d_in[2];
    const int*   mask      = (const int*)d_in[3];
    const float* conv_bias = (const float*)d_in[4];
    float* out = (float*)d_out;

    char* ws = (char*)d_ws;
    __bf16* Abf = (__bf16*)ws;                                   // 32 MB
    __bf16* Wbf = (__bf16*)(ws + (size_t)MDIM * KDIM * 2);       // 32 MB
    int*    bm  = (int*)(ws + (size_t)MDIM * KDIM * 2
                            + (size_t)NDIM * KDIM * 2);          // 64 KB

    block_mask_kernel<<<NH * NW, 256, 0, stream>>>(mask, conv_bias, bm);
    cvt_data_kernel<<<(MDIM * (size_t)KDIM / 4) / 256, 256, 0, stream>>>(data, Abf);
    cvt_weight_kernel<<<(NDIM * (size_t)KDIM / 4) / 256, 256, 0, stream>>>(weight, bm, Wbf);

    dim3 grid(NDIM / BN, MDIM / BM);
    gemm_bf16_wmma_kernel<<<grid, 256, 0, stream>>>(Abf, Wbf, bias, bm, out);
}
